// RNN_42666205118776
// MI455X (gfx1250) — matrix operand order repaired and checked by reading, not compiled, not run on hardware
//
#include <hip/hip_runtime.h>

// ---------------------------------------------------------------------------
// Continuous-time RNN, 500 sequential Euler steps (batch-independent).
//   h' = h + a*(-h + tanh(h) @ Wrec^T + u @ Winp^T + noise*std)
// 8 workgroups x 16 batch rows; h lives in WMMA f32 accumulators across the
// whole time loop. Matmul: v_wmma_f32_16x16x32_f16.
//   A = tanh(h_t) f16 in LDS (rebuilt each step)
//   B = Wrec^T f16: K rows 0..255 resident in LDS (CDNA5: 320KB/WGP),
//       K rows 256..511 streamed from L2 each step.
//   out_{t-1} = tanh(h_t) @ Wout^T is FUSED as a 5th WMMA tile on wave 0
//       (same A fragments, B = Wout^T padded to [512][16] f16 in LDS).
// A laundered zero-SGPR (empty asm) makes B addresses loop-variant so the
// compiler cannot hoist 512 VGPRs of B fragments and spill them to scratch
// (which is what happened in the previous round).
// ---------------------------------------------------------------------------

constexpr int N_REC = 512;
constexpr int N_INP = 3;
constexpr int N_OUT = 3;
constexpr int BATCH = 128;
constexpr int SEQ   = 500;
constexpr float ALPHA     = 0.1f;
constexpr float DECAY_OVER_ALPHA = (1.0f - ALPHA) / ALPHA;   // 9.0
constexpr float NOISE_STD = 0.05f;

constexpr int M_TILE = 16;          // batch rows per workgroup
constexpr int RS     = N_REC + 8;   // f16 row stride: 1040B (16B aligned, 260 dw)
constexpr int K_LDS  = 256;         // K rows of Wrec^T resident in LDS
constexpr int WOS    = 24;          // Wout^T LDS row stride (48B: 16B aligned)

typedef _Float16 v16h __attribute__((ext_vector_type(16)));
typedef _Float16 v8h  __attribute__((ext_vector_type(8)));
typedef float    v8f  __attribute__((ext_vector_type(8)));

__device__ __forceinline__ float fast_tanh(float x) {
    // tanh(x) = 1 - 2/(exp(2x)+1): v_exp_f32 + v_rcp_f32 trans ops,
    // co-execute with 16-bit WMMA. Saturates to +/-1 for large |x|.
    float e = __expf(2.0f * x);
    return 1.0f - 2.0f * __builtin_amdgcn_rcpf(e + 1.0f);
}

// One-time prep: WT[k][n] = (f16) w_rec[n][k]
__global__ void wrec_transpose_f16(const float* __restrict__ w,
                                   _Float16* __restrict__ wt) {
    __shared__ float tile[32][33];
    const int bx = blockIdx.x * 32;   // k block
    const int by = blockIdx.y * 32;   // n block
    for (int r = threadIdx.y; r < 32; r += blockDim.y)
        tile[r][threadIdx.x] = w[(size_t)(by + r) * N_REC + (bx + threadIdx.x)];
    __syncthreads();
    for (int r = threadIdx.y; r < 32; r += blockDim.y)
        wt[(size_t)(bx + r) * N_REC + (by + threadIdx.x)] =
            (_Float16)tile[threadIdx.x][r];
}

// One-time prep: WOT[k][j] = (f16) w_out[j][k], padded to 16 columns
__global__ void wout_pad_f16(const float* __restrict__ w_out,
                             _Float16* __restrict__ wot) {
    const int k = blockIdx.x * blockDim.x + threadIdx.x;
    if (k < N_REC) {
        #pragma unroll
        for (int j = 0; j < 16; ++j)
            wot[k * 16 + j] =
                (j < N_OUT) ? (_Float16)w_out[j * N_REC + k] : (_Float16)0.0f;
    }
}

__global__ __launch_bounds__(256, 1)
void rnn_persistent(const float* __restrict__ inp,      // [B][S][3]
                    const float* __restrict__ noise,    // [B][S][512]
                    const float* __restrict__ x0,       // [1][512]
                    const float* __restrict__ w_inp,    // [512][3]
                    const _Float16* __restrict__ WT,    // [512][512] = Wrec^T
                    const _Float16* __restrict__ WOT,   // [512][16]  = Wout^T pad
                    float* __restrict__ hidden,         // [B][S+1][512]
                    float* __restrict__ outputs)        // [B][S][3]
{
    __shared__ _Float16 B_lds[K_LDS * RS];        // Wrec^T rows 0..255 (260KB)
    __shared__ _Float16 r_lds[M_TILE * RS];       // tanh(h_t) f16
    __shared__ _Float16 wot_lds[N_REC * WOS];     // Wout^T padded (24KB)
    __shared__ float    winp_lds[N_REC * N_INP];
    __shared__ float    u_lds[M_TILE * N_INP];

    const int tid   = threadIdx.x;
    const int wave  = tid >> 5;                   // 8 waves (wave32)
    const int lane  = tid & 31;
    const int lhalf = lane & 15;                  // N col in 16x16 tile / A row
    const int mbase = (lane < 16) ? 0 : 8;        // D-layout row base
    const int b0    = blockIdx.x * M_TILE;
    const int nbase = wave * 64;                  // this wave's N slice

    // ---- one-time LDS fills ----
    for (int r = wave; r < K_LDS; r += 8) {       // Wrec^T K-half -> LDS
        const v8h* src = (const v8h*)(WT + (size_t)r * N_REC);
        v8h*       dst = (v8h*)(B_lds + r * RS);
        dst[lane]      = src[lane];
        dst[lane + 32] = src[lane + 32];
    }
    for (int r = tid; r < N_REC; r += 256) {      // Wout^T -> LDS (2x v8h, 16B ok)
        const v8h* src = (const v8h*)(WOT + r * 16);
        v8h*       dst = (v8h*)(wot_lds + r * WOS);
        dst[0] = src[0];
        dst[1] = src[1];
    }
    for (int i = tid; i < N_REC * N_INP; i += 256) winp_lds[i] = w_inp[i];

    // h_0 = broadcast(x0) in accumulator (D-matrix) layout
    v8f acc[4];
    #pragma unroll
    for (int nt = 0; nt < 4; ++nt) {
        float xv = x0[nbase + nt * 16 + lhalf];
        #pragma unroll
        for (int v = 0; v < 8; ++v) acc[nt][v] = xv;
    }
    __syncthreads();

    for (int t = 0; t <= SEQ; ++t) {
        // --- r_t = tanh(h_t) -> LDS f16; store hidden[:,t,:] ---
        #pragma unroll
        for (int nt = 0; nt < 4; ++nt) {
            const int n = nbase + nt * 16 + lhalf;
            #pragma unroll
            for (int v = 0; v < 8; ++v) {
                const int m = mbase + v;
                const float h = acc[nt][v];
                r_lds[m * RS + n] = (_Float16)fast_tanh(h);
                hidden[((size_t)(b0 + m) * (SEQ + 1) + t) * N_REC + n] = h;
            }
        }
        if (t < SEQ && tid < M_TILE * N_INP) {
            const int m = tid / N_INP, i = tid % N_INP;
            u_lds[tid] = inp[((size_t)(b0 + m) * SEQ + t) * N_INP + i];
        }
        __syncthreads();

        // laundered zero: blocks LICM of B-fragment loads (no scratch spills)
        int zero = 0;
        asm volatile("" : "+s"(zero));

        if (t < SEQ) {
            // --- input projection + scaled noise; prefetch next-step noise ---
            float inpn[4][8];
            #pragma unroll
            for (int nt = 0; nt < 4; ++nt) {
                const int n = nbase + nt * 16 + lhalf;
                const float wi0 = winp_lds[n * N_INP + 0];
                const float wi1 = winp_lds[n * N_INP + 1];
                const float wi2 = winp_lds[n * N_INP + 2];
                #pragma unroll
                for (int v = 0; v < 8; ++v) {
                    const int m = mbase + v;
                    const size_t row = ((size_t)(b0 + m) * SEQ + t) * N_REC;
                    float s = u_lds[m * N_INP + 0] * wi0
                            + u_lds[m * N_INP + 1] * wi1
                            + u_lds[m * N_INP + 2] * wi2;
                    s += NOISE_STD * noise[row + n];
                    if ((nt & 1) == 0 && t + 1 < SEQ)
                        __builtin_prefetch(&noise[row + N_REC + n], 0, 1);
                    inpn[nt][v] = s;
                }
            }

            // seed C = ((1-a)/a)*h so h_new = a*(wmma_chain + inpn)
            #pragma unroll
            for (int nt = 0; nt < 4; ++nt)
                #pragma unroll
                for (int v = 0; v < 8; ++v) acc[nt][v] *= DECAY_OVER_ALPHA;

            v8f oacc = {};   // wave 0: fused out-tile accumulator

            // --- 16 K-steps x (4 + 1) wmma 16x16x32 f16 ---
            for (int kb = 0; kb < N_REC / 32; ++kb) {
                // A fragment (ISA 16-bit A 16x32 layout)
                union { v16h v; v8h h[2]; } A;
                const _Float16* ar =
                    r_lds + lhalf * RS + kb * 32 + ((lane < 16) ? 0 : 8) + zero;
                A.h[0] = *(const v8h*)(ar);
                A.h[1] = *(const v8h*)(ar + 16);

                const int k = kb * 32 + mbase;  // B k base for this lane half (table order)
                if (kb < K_LDS / 32) {
                    // B from LDS (per-element k gather, column n = lhalf)
                    const _Float16* brow = B_lds + k * RS + nbase + lhalf + zero;
                    #pragma unroll
                    for (int nt = 0; nt < 4; ++nt) {
                        union { v16h v; v8h h[2]; } B;
                        #pragma unroll
                        for (int s = 0; s < 16; ++s)
                            B.v[s] = brow[((s & 7) + (s >> 3) * 16) * RS + nt * 16];
                        acc[nt] = __builtin_amdgcn_wmma_f32_16x16x32_f16(
                            false, A.v, false, B.v, (short)0, acc[nt],
                            false, false);
                    }
                } else {
                    // B streamed from global (L2-resident WT)
                    const _Float16* brow =
                        WT + (size_t)k * N_REC + nbase + lhalf + zero;
                    #pragma unroll
                    for (int nt = 0; nt < 4; ++nt) {
                        v16h B;
                        #pragma unroll
                        for (int s = 0; s < 16; ++s)
                            B[s] = brow[(size_t)((s & 7) + (s >> 3) * 16) * N_REC + nt * 16];
                        acc[nt] = __builtin_amdgcn_wmma_f32_16x16x32_f16(
                            false, A.v, false, B, (short)0, acc[nt],
                            false, false);
                    }
                }

                if (wave == 0) {   // fused out_{t-1} tile (same A fragments)
                    const _Float16* wrow = wot_lds + k * WOS + lhalf + zero;
                    union { v16h v; v8h h[2]; } Bo;
                    #pragma unroll
                    for (int s = 0; s < 16; ++s)
                        Bo.v[s] = wrow[((s & 7) + (s >> 3) * 16) * WOS];
                    oacc = __builtin_amdgcn_wmma_f32_16x16x32_f16(
                        false, A.v, false, Bo.v, (short)0, oacc, false, false);
                }
            }

            if (wave == 0 && t >= 1 && lhalf < N_OUT) {
                const int j = lhalf;   // lane<16: rows 0..7, lane>=16: rows 8..15
                #pragma unroll
                for (int v = 0; v < 8; ++v)
                    outputs[((size_t)(b0 + mbase + v) * SEQ + (t - 1)) * N_OUT
                            + j] = oacc[v];
            }

            // --- h_{t+1} = a*(9h + rec + inpn) ---
            #pragma unroll
            for (int nt = 0; nt < 4; ++nt)
                #pragma unroll
                for (int v = 0; v < 8; ++v)
                    acc[nt][v] = ALPHA * (acc[nt][v] + inpn[nt][v]);
        } else {
            // t == SEQ: epilogue, out_{SEQ-1} = r_SEQ @ Wout^T (wave 0 only)
            if (wave == 0) {
                v8f oacc = {};
                for (int kb = 0; kb < N_REC / 32; ++kb) {
                    union { v16h v; v8h h[2]; } A;
                    const _Float16* ar = r_lds + lhalf * RS + kb * 32
                                       + ((lane < 16) ? 0 : 8) + zero;
                    A.h[0] = *(const v8h*)(ar);
                    A.h[1] = *(const v8h*)(ar + 16);
                    const _Float16* wrow =
                        wot_lds + (kb * 32 + mbase) * WOS + lhalf + zero;
                    union { v16h v; v8h h[2]; } Bo;
                    #pragma unroll
                    for (int s = 0; s < 16; ++s)
                        Bo.v[s] = wrow[((s & 7) + (s >> 3) * 16) * WOS];
                    oacc = __builtin_amdgcn_wmma_f32_16x16x32_f16(
                        false, A.v, false, Bo.v, (short)0, oacc, false, false);
                }
                if (lhalf < N_OUT) {
                    #pragma unroll
                    for (int v = 0; v < 8; ++v)
                        outputs[((size_t)(b0 + mbase + v) * SEQ + (SEQ - 1))
                                * N_OUT + lhalf] = oacc[v];
                }
            }
        }
        __syncthreads();   // all r_lds readers done before next overwrite
    }
}

extern "C" void kernel_launch(void* const* d_in, const int* in_sizes, int n_in,
                              void* d_out, int out_size, void* d_ws, size_t ws_size,
                              hipStream_t stream) {
    const float* inp    = (const float*)d_in[0];
    const float* noise  = (const float*)d_in[1];
    const float* x0     = (const float*)d_in[2];
    const float* w_inp  = (const float*)d_in[3];
    const float* w_rec  = (const float*)d_in[4];
    const float* w_out  = (const float*)d_in[5];

    float* hidden  = (float*)d_out;
    float* outputs = hidden + (size_t)BATCH * (SEQ + 1) * N_REC;

    _Float16* WT  = (_Float16*)d_ws;               // 512 KB
    _Float16* WOT = WT + (size_t)N_REC * N_REC;    // +16 KB

    dim3 tb(32, 8), tg(N_REC / 32, N_REC / 32);
    wrec_transpose_f16<<<tg, tb, 0, stream>>>(w_rec, WT);
    wout_pad_f16<<<2, 256, 0, stream>>>(w_out, WOT);

    rnn_persistent<<<BATCH / M_TILE, 256, 0, stream>>>(
        inp, noise, x0, w_inp, WT, WOT, hidden, outputs);
}
